// QuanvolutionGen200_65481071398986
// MI455X (gfx1250) — matrix operand order repaired and checked by reading, not compiled, not run on hardware
//
#include <hip/hip_runtime.h>
#include <hip/hip_fp16.h>
#include <math.h>

typedef __attribute__((ext_vector_type(16))) _Float16 v16h;
typedef __attribute__((ext_vector_type(8)))  _Float16 v8h;
typedef __attribute__((ext_vector_type(8)))  float    v8f;

#define BATCH 1024
#define KDIM  784
#define KPAD  800   // 25 k-tiles of 32
#define NCLS  10

// ---------------- 4-qubit statevector helpers (state in registers) ----------------
// index i = q0*8 + q1*4 + q2*2 + q3 ; wire w acts on bit (8 >> w)

__device__ __forceinline__ void apply_ry(float (&sr)[16], float (&si)[16],
                                         float half_theta, int bit) {
  float c = __cosf(half_theta), s = __sinf(half_theta);
  #pragma unroll
  for (int i = 0; i < 16; ++i) {
    if (i & bit) continue;
    int j = i | bit;
    float a0r = sr[i], a0i = si[i], a1r = sr[j], a1i = si[j];
    sr[i] = c * a0r - s * a1r;  si[i] = c * a0i - s * a1i;
    sr[j] = s * a0r + c * a1r;  si[j] = s * a0i + c * a1i;
  }
}

__device__ __forceinline__ void apply_rz(float (&sr)[16], float (&si)[16],
                                         float half_phi, int bit) {
  float c = __cosf(half_phi), s = __sinf(half_phi);
  #pragma unroll
  for (int i = 0; i < 16; ++i) {
    float r = sr[i], im = si[i];
    if (i & bit) {            // multiply by (c + i s)
      sr[i] = c * r - s * im;  si[i] = c * im + s * r;
    } else {                  // multiply by (c - i s)
      sr[i] = c * r + s * im;  si[i] = c * im - s * r;
    }
  }
}

__device__ __forceinline__ void cnot(float (&sr)[16], float (&si)[16],
                                     int cbit, int tbit) {
  #pragma unroll
  for (int i = 0; i < 16; ++i) {
    if ((i & cbit) && !(i & tbit)) {
      int j = i | tbit;
      float tr = sr[i], ti = si[i];
      sr[i] = sr[j]; si[i] = si[j];
      sr[j] = tr;    si[j] = ti;
    }
  }
}

// ------------- Kernel 1: fused patch extraction + quanvolution -> f16 feats -------
// featsH layout: [BATCH][KPAD] halves; k = p*4 + w for p<196, zero-padded 784..799.
__global__ __launch_bounds__(256) void quanv_kernel(const float* __restrict__ x,
                                                    const float* __restrict__ qp,
                                                    _Float16* __restrict__ featsH) {
  int gid = blockIdx.x * blockDim.x + threadIdx.x;
  if (gid >= BATCH * 200) return;
  int b = gid / 200, p = gid % 200;
  _Float16* frow = featsH + (size_t)b * KPAD + p * 4;
  if (p >= 196) {               // zero K-padding region (k = 784..799)
    frow[0] = (_Float16)0.f; frow[1] = (_Float16)0.f;
    frow[2] = (_Float16)0.f; frow[3] = (_Float16)0.f;
    return;
  }
  int r = p / 14, c = p % 14;
  const float* xb = x + (size_t)b * 784;
  float th[4];
  th[0] = xb[(2 * r) * 28 + 2 * c];
  th[1] = xb[(2 * r) * 28 + 2 * c + 1];
  th[2] = xb[(2 * r + 1) * 28 + 2 * c];
  th[3] = xb[(2 * r + 1) * 28 + 2 * c + 1];

  float sr[16], si[16];
  #pragma unroll
  for (int i = 0; i < 16; ++i) { sr[i] = 0.f; si[i] = 0.f; }
  sr[0] = 1.f;

  #pragma unroll
  for (int layer = 0; layer < 3; ++layer) {
    #pragma unroll
    for (int w = 0; w < 4; ++w) apply_ry(sr, si, 0.5f * th[w], 8 >> w);
    cnot(sr, si, 8, 4);   // (0,1)
    cnot(sr, si, 4, 2);   // (1,2)
    cnot(sr, si, 2, 1);   // (2,3)
    cnot(sr, si, 1, 8);   // (3,0)
    #pragma unroll
    for (int w = 0; w < 4; ++w) {
      const float* g = qp + (layer * 4 + w) * 3;
      apply_rz(sr, si, 0.5f * g[0], 8 >> w);
      apply_ry(sr, si, 0.5f * g[1], 8 >> w);
      apply_rz(sr, si, 0.5f * g[2], 8 >> w);
    }
  }

  float ez0 = 0.f, ez1 = 0.f, ez2 = 0.f, ez3 = 0.f;
  #pragma unroll
  for (int i = 0; i < 16; ++i) {
    float pr = sr[i] * sr[i] + si[i] * si[i];
    ez0 += (i & 8) ? -pr : pr;
    ez1 += (i & 4) ? -pr : pr;
    ez2 += (i & 2) ? -pr : pr;
    ez3 += (i & 1) ? -pr : pr;
  }
  frow[0] = (_Float16)ez0;
  frow[1] = (_Float16)ez1;
  frow[2] = (_Float16)ez2;
  frow[3] = (_Float16)ez3;
}

// ------------- Kernel 2: build padded f16 W copy (16 x KPAD) ----------------------
__global__ __launch_bounds__(256) void prep_w_kernel(const float* __restrict__ W,
                                                     _Float16* __restrict__ Wh) {
  int t = blockIdx.x * blockDim.x + threadIdx.x;
  if (t >= 16 * KPAD) return;
  int n = t / KPAD, k = t % KPAD;
  Wh[t] = (n < NCLS && k < KDIM) ? (_Float16)W[n * KDIM + k] : (_Float16)0.f;
}

// ------------- Kernel 3: WMMA GEMM (1024x784 @ 784x16) + bias + log_softmax -------
// One wave32 per 16-row tile. C/D layout: lane L holds row m = v + (L>=16?8:0),
// col n = L & 15, for v in 0..7.
__global__ __launch_bounds__(32) void gemm_lsm_kernel(const _Float16* __restrict__ featsH,
                                                      const _Float16* __restrict__ Wh,
                                                      const float* __restrict__ bias,
                                                      float* __restrict__ out) {
  int lane   = threadIdx.x;
  int laneLo = lane & 15;
  int laneHi = lane >> 4;
  int row0   = blockIdx.x * 16;

  const _Float16* arow = featsH + (size_t)(row0 + laneLo) * KPAD;
  const _Float16* brow = Wh + (size_t)laneLo * KPAD;   // column n = laneLo of W^T

  v8f acc = {};
  for (int kt = 0; kt < KPAD / 32; ++kt) {
    int k0 = kt * 32;
    // A fragment: 16 contiguous K values per lane-half (matches B)
    v8h alo = *(const v8h*)(arow + k0 + laneHi * 16);       // halves 0..7
    v8h ahi = *(const v8h*)(arow + k0 + laneHi * 16 + 8);   // halves 8..15
    v16h a;
    #pragma unroll
    for (int t = 0; t < 8; ++t) { a[t] = alo[t]; a[8 + t] = ahi[t]; }
    // B fragment: 16 contiguous K values per lane-half
    v16h bf = *(const v16h*)(brow + k0 + laneHi * 16);
    acc = __builtin_amdgcn_wmma_f32_16x16x32_f16(
        /*neg_a=*/false, a, /*neg_b=*/false, bf,
        /*c_mod=*/(short)0, acc, /*reuse_a=*/false, /*reuse_b=*/false);
  }

  float bn = (laneLo < NCLS) ? bias[laneLo] : 0.f;
  #pragma unroll
  for (int v = 0; v < 8; ++v) {
    float logit = (laneLo < NCLS) ? (acc[v] + bn) : -3.0e38f;
    // row max across the 16 lanes sharing laneHi
    float m = logit;
    #pragma unroll
    for (int mask = 1; mask < 16; mask <<= 1)
      m = fmaxf(m, __shfl_xor(m, mask, 32));
    float e = __expf(logit - m);       // invalid lanes -> exp(-inf) = 0
    float s = e;
    #pragma unroll
    for (int mask = 1; mask < 16; mask <<= 1)
      s += __shfl_xor(s, mask, 32);
    float lsm = logit - m - __logf(s);
    int row = row0 + laneHi * 8 + v;
    if (laneLo < NCLS) out[row * NCLS + laneLo] = lsm;
  }
}

// ----------------------------------------------------------------------------------
extern "C" void kernel_launch(void* const* d_in, const int* in_sizes, int n_in,
                              void* d_out, int out_size, void* d_ws, size_t ws_size,
                              hipStream_t stream) {
  (void)in_sizes; (void)n_in; (void)out_size; (void)ws_size;
  const float* x    = (const float*)d_in[0];   // (1024,1,28,28)
  const float* qp   = (const float*)d_in[1];   // (3,4,3)
  const float* W    = (const float*)d_in[2];   // (10,784)
  const float* bias = (const float*)d_in[3];   // (10,)
  float* out = (float*)d_out;                  // (1024,10)

  _Float16* featsH = (_Float16*)d_ws;                                    // 1024*800 halves
  _Float16* Wh     = (_Float16*)((char*)d_ws + (size_t)BATCH * KPAD * 2); // 16*800 halves

  hipLaunchKernelGGL(prep_w_kernel, dim3((16 * KPAD + 255) / 256), dim3(256), 0, stream,
                     W, Wh);
  hipLaunchKernelGGL(quanv_kernel, dim3((BATCH * 200 + 255) / 256), dim3(256), 0, stream,
                     x, qp, featsH);
  hipLaunchKernelGGL(gemm_lsm_kernel, dim3(BATCH / 16), dim3(32), 0, stream,
                     featsH, Wh, bias, out);
}
